// FlashAttention_87290915324346
// MI455X (gfx1250) — compile-verified, matrix operand order checked by reading, not run on hardware
//
#include <hip/hip_runtime.h>

#define BB 2
#define SS 2048
#define DD 1024
#define HH 16
#define HD 64
#define CHUNK 1024

typedef __attribute__((ext_vector_type(16))) _Float16 v16h;
typedef __attribute__((ext_vector_type(8)))  _Float16 v8h;
typedef __attribute__((ext_vector_type(4)))  _Float16 v4h;
typedef __attribute__((ext_vector_type(8)))  float    v8f;

#define WMMA(a, b, c) __builtin_amdgcn_wmma_f32_16x16x32_f16(false, (a), false, (b), (short)0, (c), false, false)

// Load a 16x32 f16 WMMA fragment (A layout, or B when source is stored [N,K]
// row-major). Two aligned 128-bit loads per lane (compiler merges to b128 pair).
static __device__ __forceinline__ v16h load_frag(const _Float16* __restrict__ p,
                                                 int ld, int row0, int k0, int lane) {
  int m = lane & 15;
  int koff = k0 + ((lane >> 4) << 3);
  const _Float16* base = p + (size_t)(row0 + m) * ld + koff;
  v8h lo = *(const v8h*)(base);
  v8h hi = *(const v8h*)(base + 16);
  v16h r;
#pragma unroll
  for (int i = 0; i < 8; ++i) { r[i] = lo[i]; r[i + 8] = hi[i]; }
  return r;
}

// ---------------- fp32 -> f16 convert (vector x4) ----------------
__global__ __launch_bounds__(256) void cvt_f16_kernel(const float* __restrict__ src,
                                                      _Float16* __restrict__ dst, int n4) {
  int i = blockIdx.x * blockDim.x + threadIdx.x;
  if (i >= n4) return;
  float4 v = ((const float4*)src)[i];
  v4h o = {(_Float16)v.x, (_Float16)v.y, (_Float16)v.z, (_Float16)v.w};
  ((v4h*)dst)[i] = o;
}

// ---------------- weight transpose: W[K,N] f32 -> Wt[N,K] f16 ----------------
__global__ __launch_bounds__(256) void transpose_w_kernel(const float* __restrict__ W,
                                                          _Float16* __restrict__ Wt) {
  __shared__ float tile[16][17];
  int n = blockIdx.x * 16 + threadIdx.x;
  int k = blockIdx.y * 16 + threadIdx.y;
  tile[threadIdx.y][threadIdx.x] = W[(size_t)k * DD + n];
  __syncthreads();
  int nn = blockIdx.x * 16 + threadIdx.y;
  int kk = blockIdx.y * 16 + threadIdx.x;
  Wt[(size_t)nn * DD + kk] = (_Float16)tile[threadIdx.x][threadIdx.y];
}

// ---------------- WMMA GEMM core: 32x64 wave tile ---------------------------
// acc[mt][nt] computed in-place; A [M,K] f16 row-major, Bt [N,K] f16 row-major.
template <typename STORE>
static __device__ __forceinline__ void gemm_wave_32x64(const _Float16* __restrict__ A,
                                                       const _Float16* __restrict__ Bt,
                                                       int M, int N, int K,
                                                       int wid, int lane, STORE store) {
  int ntiles = N >> 6;
  int m0 = (wid / ntiles) << 5;
  int n0 = (wid % ntiles) << 6;
  if (m0 >= M) return;
  v8f acc[2][4];
#pragma unroll
  for (int mt = 0; mt < 2; ++mt)
#pragma unroll
    for (int nt = 0; nt < 4; ++nt) acc[mt][nt] = (v8f){};

  for (int k0 = 0; k0 < K; k0 += 32) {
    // gfx1250 prefetch path (global_prefetch_b8) one macro-step ahead
    if (k0 + 256 < K) {
      __builtin_prefetch(A + (size_t)(m0 + (lane & 15)) * K + k0 + 256, 0, 0);
      __builtin_prefetch(Bt + (size_t)(n0 + (lane & 15)) * K + k0 + 256, 0, 0);
    }
    v16h a0 = load_frag(A, K, m0, k0, lane);
    v16h a1 = load_frag(A, K, m0 + 16, k0, lane);
#pragma unroll
    for (int nt = 0; nt < 4; ++nt) {
      v16h b = load_frag(Bt, K, n0 + (nt << 4), k0, lane);
      acc[0][nt] = WMMA(a0, b, acc[0][nt]);
      acc[1][nt] = WMMA(a1, b, acc[1][nt]);
    }
  }
#pragma unroll
  for (int mt = 0; mt < 2; ++mt) {
    int rbase = m0 + (mt << 4) + ((lane >> 4) << 3);
    int cbase = n0 + (lane & 15);
#pragma unroll
    for (int nt = 0; nt < 4; ++nt)
#pragma unroll
      for (int r = 0; r < 8; ++r)
        store(rbase + r, cbase + (nt << 4), acc[mt][nt][r]);
  }
}

// layout: 0 = row-major [M,N]; 1 = [B,H,S,HD]; 2 = [B,H,HD,S] (transposed V)
__global__ __launch_bounds__(256) void gemm_f16out_kernel(const _Float16* __restrict__ A,
                                                          const _Float16* __restrict__ Bt,
                                                          _Float16* __restrict__ C,
                                                          int M, int N, int K, int layout) {
  int wid  = (blockIdx.x * blockDim.x + threadIdx.x) >> 5;
  int lane = threadIdx.x & 31;
  gemm_wave_32x64(A, Bt, M, N, K, wid, lane, [&](int row, int col, float v) {
    size_t idx;
    if (layout == 0) {
      idx = (size_t)row * N + col;
    } else {
      int b = row >> 11, s = row & (SS - 1);
      int h = col >> 6, d = col & 63;
      if (layout == 1) idx = (((size_t)(b * HH + h) * SS + s) << 6) + d;
      else             idx = (((size_t)(b * HH + h) * HD + d) * SS) + s;
    }
    C[idx] = (_Float16)v;
  });
}

__global__ __launch_bounds__(256) void gemm_f32out_kernel(const _Float16* __restrict__ A,
                                                          const _Float16* __restrict__ Bt,
                                                          float* __restrict__ C,
                                                          int M, int N, int K) {
  int wid  = (blockIdx.x * blockDim.x + threadIdx.x) >> 5;
  int lane = threadIdx.x & 31;
  gemm_wave_32x64(A, Bt, M, N, K, wid, lane, [&](int row, int col, float v) {
    C[(size_t)row * N + col] = v;
  });
}

// ---------------- RoPE over Q,K in [B,H,S,HD] f16 --------------------------
__global__ __launch_bounds__(256) void rope_kernel(_Float16* __restrict__ Q,
                                                   _Float16* __restrict__ K,
                                                   const float* __restrict__ fc) {
  size_t idx = (size_t)blockIdx.x * blockDim.x + threadIdx.x;  // B*H*S*32
  int j = (int)(idx & 31);
  size_t bhs = idx >> 5;
  int s = (int)(bhs & (SS - 1));
  float c = fc[(s * 32 + j) * 2 + 0];
  float d = fc[(s * 32 + j) * 2 + 1];
  size_t base = (bhs << 6) + ((size_t)j << 1);
  float a = (float)Q[base], b = (float)Q[base + 1];
  Q[base]     = (_Float16)(a * c - b * d);
  Q[base + 1] = (_Float16)(a * d + b * c);
  a = (float)K[base]; b = (float)K[base + 1];
  K[base]     = (_Float16)(a * c - b * d);
  K[base + 1] = (_Float16)(a * d + b * c);
}

// ---------------- attention: per-chunk softmax, chunk outputs summed -------
__global__ __launch_bounds__(128) void attn_kernel(const _Float16* __restrict__ Q,
                                                   const _Float16* __restrict__ Kd,
                                                   const _Float16* __restrict__ Vt,
                                                   _Float16* __restrict__ O) {
  __shared__ __align__(16) _Float16 pbuf[4][16 * 32];  // per-wave P tile
  int wid    = (blockIdx.x * blockDim.x + threadIdx.x) >> 5;
  int wlocal = threadIdx.x >> 5;
  int lane   = threadIdx.x & 31;
  int qt = wid & ((SS / 16) - 1);
  int bh = wid >> 7;
  const _Float16* Qb = Q  + (size_t)bh * SS * HD;
  const _Float16* Kb = Kd + (size_t)bh * SS * HD;
  const _Float16* Vb = Vt + (size_t)bh * HD * SS;
  int m0 = qt << 4;

  v16h qa0 = load_frag(Qb, HD, m0, 0,  lane);
  v16h qa1 = load_frag(Qb, HD, m0, 32, lane);

  float ofin[4][8];
#pragma unroll
  for (int t = 0; t < 4; ++t)
#pragma unroll
    for (int r = 0; r < 8; ++r) ofin[t][r] = 0.f;

  _Float16* pl = pbuf[wlocal];
  const float scale = 0.125f;  // 64^-0.5

  for (int c0 = 0; c0 < SS; c0 += CHUNK) {
    float mrun[8], lrun[8];
#pragma unroll
    for (int r = 0; r < 8; ++r) { mrun[r] = -1e30f; lrun[r] = 0.f; }
    v8f oa0 = {}, oa1 = {}, oa2 = {}, oa3 = {};

    for (int kk = 0; kk < CHUNK; kk += 32) {
      // ---- scores: 16 queries x 32 keys, K-dim = HD = 64 (2 steps) ----
      v8f s0 = {}, s1 = {};
      {
        v16h kb = load_frag(Kb, HD, c0 + kk, 0, lane);
        s0 = WMMA(qa0, kb, s0);
        kb = load_frag(Kb, HD, c0 + kk, 32, lane);
        s0 = WMMA(qa1, kb, s0);
        kb = load_frag(Kb, HD, c0 + kk + 16, 0, lane);
        s1 = WMMA(qa0, kb, s1);
        kb = load_frag(Kb, HD, c0 + kk + 16, 32, lane);
        s1 = WMMA(qa1, kb, s1);
      }
      // ---- online softmax (rows live across 16-lane half-waves) ----
      float alpha[8];
#pragma unroll
      for (int r = 0; r < 8; ++r) {
        float v0 = s0[r] * scale, v1 = s1[r] * scale;
        float mx = fmaxf(v0, v1);
#pragma unroll
        for (int off = 1; off < 16; off <<= 1)
          mx = fmaxf(mx, __shfl_xor(mx, off, 32));
        float mnew = fmaxf(mrun[r], mx);
        alpha[r] = __expf(mrun[r] - mnew);
        float p0 = __expf(v0 - mnew);
        float p1 = __expf(v1 - mnew);
        float ps = p0 + p1;
#pragma unroll
        for (int off = 1; off < 16; off <<= 1)
          ps += __shfl_xor(ps, off, 32);
        lrun[r] = lrun[r] * alpha[r] + ps;
        mrun[r] = mnew;
        s0[r] = p0;
        s1[r] = p1;
      }
#pragma unroll
      for (int r = 0; r < 8; ++r) {
        oa0[r] *= alpha[r]; oa1[r] *= alpha[r];
        oa2[r] *= alpha[r]; oa3[r] *= alpha[r];
      }
      // ---- reshape P: C-layout -> A-fragment layout via LDS ----
      {
        int rb = (lane >> 4) << 3;
        int cc = lane & 15;
#pragma unroll
        for (int r = 0; r < 8; ++r) {
          pl[(rb + r) * 32 + cc]      = (_Float16)s0[r];
          pl[(rb + r) * 32 + 16 + cc] = (_Float16)s1[r];
        }
      }
      asm volatile("s_wait_dscnt 0x0" ::: "memory");
      v16h pa;
      {
        int m = lane & 15;
        int koff = (lane >> 4) << 3;
        const _Float16* pb = pl + m * 32 + koff;
        v8h lo = *(const v8h*)pb;
        v8h hi = *(const v8h*)(pb + 16);
#pragma unroll
        for (int i = 0; i < 8; ++i) { pa[i] = lo[i]; pa[i + 8] = hi[i]; }
      }
      // ---- P @ V: K-dim = 32 keys, 4 d-tiles (Vt is [HD,S]) ----
      v16h vb = load_frag(Vb, SS, 0, c0 + kk, lane);
      oa0 = WMMA(pa, vb, oa0);
      vb = load_frag(Vb, SS, 16, c0 + kk, lane);
      oa1 = WMMA(pa, vb, oa1);
      vb = load_frag(Vb, SS, 32, c0 + kk, lane);
      oa2 = WMMA(pa, vb, oa2);
      vb = load_frag(Vb, SS, 48, c0 + kk, lane);
      oa3 = WMMA(pa, vb, oa3);
    }
    // ---- finalize chunk (normalize by chunk-local l, accumulate) ----
#pragma unroll
    for (int r = 0; r < 8; ++r) {
      float inv = 1.f / lrun[r];
      ofin[0][r] += oa0[r] * inv;
      ofin[1][r] += oa1[r] * inv;
      ofin[2][r] += oa2[r] * inv;
      ofin[3][r] += oa3[r] * inv;
    }
  }
  // ---- store O as f16 [B*S, D] for final GEMM ----
  int b = bh >> 4, h = bh & 15;
  int rbase = m0 + ((lane >> 4) << 3);
  int cb = (h << 6) + (lane & 15);
#pragma unroll
  for (int t = 0; t < 4; ++t)
#pragma unroll
    for (int r = 0; r < 8; ++r) {
      int row = rbase + r;
      O[((size_t)(b * SS + row)) * DD + cb + (t << 4)] = (_Float16)ofin[t][r];
    }
}

extern "C" void kernel_launch(void* const* d_in, const int* in_sizes, int n_in,
                              void* d_out, int out_size, void* d_ws, size_t ws_size,
                              hipStream_t stream) {
  const float* X  = (const float*)d_in[0];
  const float* FC = (const float*)d_in[1];
  const float* Wq = (const float*)d_in[2];
  const float* Wk = (const float*)d_in[3];
  const float* Wv = (const float*)d_in[4];
  const float* Wo = (const float*)d_in[5];
  float* out = (float*)d_out;

  char* ws = (char*)d_ws;
  const size_t MB = 1u << 20;
  _Float16* Xh  = (_Float16*)(ws + 0);        // 8 MB  [B*S, D] f16
  _Float16* Wqt = (_Float16*)(ws + 8  * MB);  // 2 MB  [N,K] f16
  _Float16* Wkt = (_Float16*)(ws + 10 * MB);
  _Float16* Wvt = (_Float16*)(ws + 12 * MB);
  _Float16* Wot = (_Float16*)(ws + 14 * MB);
  _Float16* Qh  = (_Float16*)(ws + 16 * MB);  // 8 MB  [B,H,S,HD]
  _Float16* Kh  = (_Float16*)(ws + 24 * MB);  // 8 MB  [B,H,S,HD]
  _Float16* Vtb = (_Float16*)(ws + 32 * MB);  // 8 MB  [B,H,HD,S]
  _Float16* Oh  = (_Float16*)(ws + 40 * MB);  // 8 MB  [B*S, D]

  const int M = BB * SS;  // 4096

  // 1) X fp32 -> f16
  cvt_f16_kernel<<<(M * DD / 4) / 256, 256, 0, stream>>>(X, Xh, M * DD / 4);

  // 2) weights fp32 [K,N] -> f16 [N,K]
  {
    dim3 tb(16, 16), tg(DD / 16, DD / 16);
    transpose_w_kernel<<<tg, tb, 0, stream>>>(Wq, Wqt);
    transpose_w_kernel<<<tg, tb, 0, stream>>>(Wk, Wkt);
    transpose_w_kernel<<<tg, tb, 0, stream>>>(Wv, Wvt);
    transpose_w_kernel<<<tg, tb, 0, stream>>>(Wo, Wot);
  }

  // 3) QKV projections: waves = (M/32)*(N/64) = 2048 -> 256 blocks x 256 thr
  gemm_f16out_kernel<<<256, 256, 0, stream>>>(Xh, Wqt, Qh,  M, DD, DD, 1);
  gemm_f16out_kernel<<<256, 256, 0, stream>>>(Xh, Wkt, Kh,  M, DD, DD, 1);
  gemm_f16out_kernel<<<256, 256, 0, stream>>>(Xh, Wvt, Vtb, M, DD, DD, 2);

  // 4) RoPE on Q and K
  rope_kernel<<<(BB * HH * SS * 32) / 256, 256, 0, stream>>>(Qh, Kh, FC);

  // 5) attention (waves = B*H*(S/16) = 4096 -> 1024 blocks x 128 thr)
  attn_kernel<<<1024, 128, 0, stream>>>(Qh, Kh, Vtb, Oh);

  // 6) output projection -> f32 d_out
  gemm_f32out_kernel<<<256, 256, 0, stream>>>(Oh, Wot, out, M, DD, DD);
}
